// CausalSelfAttention_60138132079103
// MI455X (gfx1250) — matrix operand order repaired and checked by reading, not compiled, not run on hardware
//
#include <hip/hip_runtime.h>

// ---------------------------------------------------------------- types
typedef __attribute__((ext_vector_type(16))) __bf16 v16bf;
typedef __attribute__((ext_vector_type(8)))  float  v8f;

union BFrag {
    v16bf v;
    uint4 q[2];
};

__device__ inline v8f wmma_bf16(v16bf a, v16bf b, v8f c) {
    // D = A(16x32 bf16) x B(32x16 bf16) + C(16x16 f32)
    return __builtin_amdgcn_wmma_f32_16x16x32_bf16(
        /*neg_a=*/false, a, /*neg_b=*/false, b,
        /*c_mod=*/(short)0, c, /*reuse_a=*/false, /*reuse_b=*/false);
}

// CDNA5 async copy global->LDS (ASYNCcnt path). VDST carries the LDS byte
// address (= low 32 bits of the generic pointer, per ISA aperture rule).
__device__ inline void async_load_b128(const __bf16* lds_dst, const __bf16* gsrc) {
    unsigned lds_off = (unsigned)(unsigned long long)lds_dst;
    asm volatile("global_load_async_to_lds_b128 %0, %1, off"
                 :: "v"(lds_off), "v"(gsrc)
                 : "memory");
}
__device__ inline void wait_async0() {
    asm volatile("s_wait_asynccnt 0" ::: "memory");
}

__device__ inline float red_max16(float v) {
#pragma unroll
    for (int off = 8; off > 0; off >>= 1)
        v = fmaxf(v, __shfl_xor(v, off, 32));
    return v;
}
__device__ inline float red_sum16(float v) {
#pragma unroll
    for (int off = 8; off > 0; off >>= 1)
        v += __shfl_xor(v, off, 32);
    return v;
}

__device__ inline void store_out(float* p, float v)  { *p = v; }
__device__ inline void store_out(__bf16* p, float v) { *p = (__bf16)v; }

// ---------------------------------------------------------------- cast
__global__ void cast_f32_to_bf16(const float* __restrict__ src,
                                 __bf16* __restrict__ dst, int n) {
    int i = (blockIdx.x * blockDim.x + threadIdx.x) * 4;
    if (i + 3 < n) {
        float4 f = *(const float4*)(src + i);
        dst[i + 0] = (__bf16)f.x;
        dst[i + 1] = (__bf16)f.y;
        dst[i + 2] = (__bf16)f.z;
        dst[i + 3] = (__bf16)f.w;
    }
}

// ---------------------------------------------------------------- GEMM  C = A * W^T
// A: [M,K] bf16 row-major.  W: [N,K] bf16 row-major (torch [out,in]).
// Block = 256 thr (8 waves) computes a 128(M) x 64(N) tile; wave owns 16x64.
// A/W K-slabs (128x32 / 64x32) are double-buffered in LDS, filled with
// global_load_async_to_lds_b128 and consumed by v_wmma_f32_16x16x32_bf16.
template <typename OUT_T>
__global__ __launch_bounds__(256)
void gemm_xwt(const __bf16* __restrict__ A, const __bf16* __restrict__ W,
              OUT_T* __restrict__ C, int M, int N, int K) {
    __shared__ __bf16 Asm[2][128 * 32];
    __shared__ __bf16 Wsm[2][64 * 32];

    const int tid  = threadIdx.x;
    const int lane = tid & 31;
    const int wave = tid >> 5;
    const int mblk = blockIdx.y * 128;
    const int n0   = blockIdx.x * 64;
    const int mrow = lane & 15;   // A-frag: lanes 0-15/16-31 both hold M=0..15
    const int half = lane >> 4;   // A-frag K-half selector
    const int ncol = lane & 15;   // B/C: N across lanes 0-15
    const int grp  = lane >> 4;   // B: K group; C: M + 8

    // staging roles
    const int ra = tid >> 1, sa = tid & 1;  // A: 128 rows x (2 segs of 16 elem)
    const int rw = tid >> 2, sw = tid & 3;  // W:  64 rows x (4 segs of 8 elem)

    v8f zero = {0.f, 0.f, 0.f, 0.f, 0.f, 0.f, 0.f, 0.f};
    v8f acc[4];
#pragma unroll
    for (int t = 0; t < 4; t++) acc[t] = zero;

    const __bf16* agrow = A + (size_t)(mblk + ra) * K + sa * 16;
    const __bf16* wgrow = W + (size_t)(n0 + rw) * K + sw * 8;

    // prime buffer 0
    {
        __bf16* ad = &Asm[0][ra * 32 + sa * 16];
        async_load_b128(ad,     agrow);
        async_load_b128(ad + 8, agrow + 8);
        async_load_b128(&Wsm[0][rw * 32 + sw * 8], wgrow);
    }

    int cur = 0;
    for (int kk = 0; kk < K; kk += 32) {
        wait_async0();        // my async writes (into buf[cur]) are done
        __syncthreads();      // everyone's are; prev buffer free for reuse
        if (kk + 32 < K) {    // prefetch next K-slab into other buffer
            __bf16* ad = &Asm[cur ^ 1][ra * 32 + sa * 16];
            async_load_b128(ad,     agrow + kk + 32);
            async_load_b128(ad + 8, agrow + kk + 40);
            async_load_b128(&Wsm[cur ^ 1][rw * 32 + sw * 8], wgrow + kk + 32);
        }
        // consume buf[cur]
        BFrag a;
        {
            const __bf16* ap = &Asm[cur][(wave * 16 + mrow) * 32 + half * 8];
            a.q[0] = *(const uint4*)(ap);        // K = h*8 .. +7
            a.q[1] = *(const uint4*)(ap + 16);   // K = 16 + h*8 .. +7
        }
#pragma unroll
        for (int t = 0; t < 4; t++) {
            const __bf16* bp = &Wsm[cur][(t * 16 + ncol) * 32 + grp * 8];
            BFrag b;
            b.q[0] = *(const uint4*)(bp);
            b.q[1] = *(const uint4*)(bp + 16);
            acc[t] = wmma_bf16(a.v, b.v, acc[t]);
        }
        cur ^= 1;
    }
    // C layout: VGPR v -> M = v + 8*(lane>=16), N = lane&15
#pragma unroll
    for (int t = 0; t < 4; t++) {
#pragma unroll
        for (int v = 0; v < 8; v++) {
            int m = v + 8 * grp;
            size_t idx = (size_t)(mblk + wave * 16 + m) * N + n0 + t * 16 + ncol;
            store_out(C + idx, acc[t][v]);
        }
    }
}

// ---------------------------------------------------------------- attention
#define SEQL   2048
#define NHEAD  16

// grid: (B*H, S/128). Block 256 thr = 8 waves; wave owns 16 query rows.
__global__ __launch_bounds__(256)
void attn_kernel(const __bf16* __restrict__ Qm, const __bf16* __restrict__ Km,
                 const __bf16* __restrict__ Vm, __bf16* __restrict__ Om) {
    __shared__ __bf16 Ksm[64 * 64];          // K block, row-major [k][d]
    __shared__ __bf16 VTsm[64 * 64];         // V block transposed [d][k]
    __shared__ __bf16 Psm[8 * 16 * 64];      // per-wave P scratch

    const int bh = blockIdx.x;
    const int b = bh >> 4;
    const int h = bh & 15;
    const int q0 = blockIdx.y * 128;
    const int tid = threadIdx.x;
    const int lane = tid & 31;
    const int wave = tid >> 5;
    const int qw0 = q0 + wave * 16;

    const int mrow = lane & 15;
    const int half = lane >> 4;
    const int ncol = lane & 15;
    const int grp  = lane >> 4;

    const size_t rowbase = (size_t)(b * SEQL) * 1024 + h * 64;

    // preload Q tile (16x64) as two 16x32 A fragments
    BFrag aq[2];
    {
        const __bf16* qp = Qm + rowbase + (size_t)(qw0 + mrow) * 1024 + half * 8;
#pragma unroll
        for (int f = 0; f < 2; f++) {
            aq[f].q[0] = *(const uint4*)(qp + f * 32);
            aq[f].q[1] = *(const uint4*)(qp + f * 32 + 16);
        }
    }

    v8f zero = {0.f, 0.f, 0.f, 0.f, 0.f, 0.f, 0.f, 0.f};
    v8f o_acc[4];
#pragma unroll
    for (int t = 0; t < 4; t++) o_acc[t] = zero;
    float mrun[8], lrun[8];
#pragma unroll
    for (int v = 0; v < 8; v++) { mrun[v] = -3.0e38f; lrun[v] = 0.f; }

    __bf16* Pw = Psm + wave * 16 * 64;

    const int kb_hi = (q0 + 127) >> 6;
    for (int kb = 0; kb <= kb_hi; kb++) {
        // ---- stage K block via async-to-LDS; V block transposed manually
        {
            int r = tid >> 2, seg = tid & 3;
            const __bf16* kp = Km + rowbase + (size_t)(kb * 64 + r) * 1024 + seg * 16;
            __bf16* kd = &Ksm[r * 64 + seg * 16];
            async_load_b128(kd,     kp);        // overlaps with the V transpose
            async_load_b128(kd + 8, kp + 8);
            const __bf16* vp = Vm + rowbase + (size_t)(kb * 64 + r) * 1024 + seg * 16;
            union { uint4 q[2]; __bf16 e[16]; } vv;
            vv.q[0] = *(const uint4*)vp;
            vv.q[1] = *(const uint4*)(vp + 8);
#pragma unroll
            for (int i = 0; i < 16; i++)
                VTsm[(seg * 16 + i) * 64 + r] = vv.e[i];
            if (kb < kb_hi)  // hint next key block toward L2
                __builtin_prefetch(kp + 64 * 1024, 0, 0);
        }
        wait_async0();
        __syncthreads();

        if (kb * 64 <= qw0 + 15) {
            // ---- S = Q K^T  (4 tiles of 16 keys, 2 chained K=32 WMMAs each)
            v8f s[4];
#pragma unroll
            for (int t = 0; t < 4; t++) {
                v8f sc = zero;
#pragma unroll
                for (int f = 0; f < 2; f++) {
                    BFrag bk;
                    const __bf16* kp = &Ksm[(t * 16 + ncol) * 64 + f * 32 + grp * 8];
                    bk.q[0] = *(const uint4*)kp;
                    bk.q[1] = *(const uint4*)(kp + 16);
                    sc = wmma_bf16(aq[f].v, bk.v, sc);
                }
                s[t] = sc;
            }
            // ---- scale 1/sqrt(64) then causal mask with reference's -10000 fill
#pragma unroll
            for (int t = 0; t < 4; t++) {
#pragma unroll
                for (int v = 0; v < 8; v++) {
                    int qg = qw0 + v + 8 * grp;
                    int kg = kb * 64 + t * 16 + ncol;
                    float val = s[t][v] * 0.125f;
                    s[t][v] = (kg > qg) ? -10000.0f : val;
                }
            }
            // ---- online softmax (rows live in 16-lane groups)
#pragma unroll
            for (int v = 0; v < 8; v++) {
                float rm = fmaxf(fmaxf(s[0][v], s[1][v]), fmaxf(s[2][v], s[3][v]));
                rm = red_max16(rm);
                float mnew = fmaxf(mrun[v], rm);
                float corr = __expf(mrun[v] - mnew);
                mrun[v] = mnew;
                float rs = 0.f;
#pragma unroll
                for (int t = 0; t < 4; t++) {
                    float p = __expf(s[t][v] - mnew);
                    s[t][v] = p;
                    rs += p;
                }
                rs = red_sum16(rs);
                lrun[v] = lrun[v] * corr + rs;
#pragma unroll
                for (int t = 0; t < 4; t++) o_acc[t][v] *= corr;
            }
            // ---- P: C-layout -> row-major in wave-private LDS
#pragma unroll
            for (int t = 0; t < 4; t++)
#pragma unroll
                for (int v = 0; v < 8; v++) {
                    int m = v + 8 * grp;
                    Pw[m * 64 + t * 16 + ncol] = (__bf16)s[t][v];
                }
            // ---- reload P as A fragments
            BFrag pa[2];
            {
                const __bf16* pp = Pw + mrow * 64 + half * 8;
#pragma unroll
                for (int f = 0; f < 2; f++) {
                    pa[f].q[0] = *(const uint4*)(pp + f * 32);
                    pa[f].q[1] = *(const uint4*)(pp + f * 32 + 16);
                }
            }
            // ---- O += P V   (V^T in LDS gives contiguous B fragments)
#pragma unroll
            for (int t = 0; t < 4; t++) {
#pragma unroll
                for (int f = 0; f < 2; f++) {
                    BFrag bv;
                    const __bf16* vp2 = &VTsm[(t * 16 + ncol) * 64 + f * 32 + grp * 8];
                    bv.q[0] = *(const uint4*)vp2;
                    bv.q[1] = *(const uint4*)(vp2 + 16);
                    o_acc[t] = wmma_bf16(pa[f].v, bv.v, o_acc[t]);
                }
            }
        }
        __syncthreads();
    }

    // ---- normalize and store attention output (bf16, [B*S, 1024])
#pragma unroll
    for (int v = 0; v < 8; v++) {
        float inv = 1.0f / lrun[v];
        int m = v + 8 * grp;
        size_t row = (size_t)(b * SEQL + qw0 + m) * 1024 + h * 64;
#pragma unroll
        for (int t = 0; t < 4; t++)
            Om[row + t * 16 + ncol] = (__bf16)(o_acc[t][v] * inv);
    }
}

// ---------------------------------------------------------------- launch
extern "C" void kernel_launch(void* const* d_in, const int* in_sizes, int n_in,
                              void* d_out, int out_size, void* d_ws, size_t ws_size,
                              hipStream_t stream) {
    (void)in_sizes; (void)n_in; (void)out_size; (void)ws_size;
    const float* x  = (const float*)d_in[0];
    const float* wq = (const float*)d_in[1];
    const float* wk = (const float*)d_in[2];
    const float* wv = (const float*)d_in[3];
    const float* wo = (const float*)d_in[4];

    constexpr size_t NX = 4096ull * 1024;   // x / Q / K / V / O element count
    constexpr size_t NW = 1024ull * 1024;   // one weight matrix

    __bf16* xb  = (__bf16*)d_ws;
    __bf16* wqb = xb + NX;
    __bf16* wkb = wqb + NW;
    __bf16* wvb = wkb + NW;
    __bf16* wob = wvb + NW;
    __bf16* Qb  = wob + NW;
    __bf16* Kb  = Qb + NX;
    __bf16* Vb  = Kb + NX;
    __bf16* Ob  = Vb + NX;                  // total 48 MiB of d_ws

    cast_f32_to_bf16<<<NX / 1024, 256, 0, stream>>>(x,  xb,  (int)NX);
    cast_f32_to_bf16<<<NW / 1024, 256, 0, stream>>>(wq, wqb, (int)NW);
    cast_f32_to_bf16<<<NW / 1024, 256, 0, stream>>>(wk, wkb, (int)NW);
    cast_f32_to_bf16<<<NW / 1024, 256, 0, stream>>>(wv, wvb, (int)NW);
    cast_f32_to_bf16<<<NW / 1024, 256, 0, stream>>>(wo, wob, (int)NW);

    dim3 gg(1024 / 64, 4096 / 128);         // (N/64, M/128)
    gemm_xwt<__bf16><<<gg, 256, 0, stream>>>(xb, wqb, Qb, 4096, 1024, 1024);
    gemm_xwt<__bf16><<<gg, 256, 0, stream>>>(xb, wkb, Kb, 4096, 1024, 1024);
    gemm_xwt<__bf16><<<gg, 256, 0, stream>>>(xb, wvb, Vb, 4096, 1024, 1024);

    attn_kernel<<<dim3(2 * NHEAD, SEQL / 128), 256, 0, stream>>>(Qb, Kb, Vb, Ob);

    gemm_xwt<float><<<gg, 256, 0, stream>>>(Ob, wob, (float*)d_out, 4096, 1024, 1024);
}
